// PureTriXFFTN8_63806034149901
// MI455X (gfx1250) — compile-verified, matrix operand order checked by reading, not run on hardware
//
#include <hip/hip_runtime.h>
#include <hip/hip_bf16.h>
#include <math.h>

typedef __attribute__((ext_vector_type(16))) _Float16 v16h;
typedef __attribute__((ext_vector_type(8)))  _Float16 v8h;
typedef __attribute__((ext_vector_type(8)))  float    v8f;

#define NV_ 8
#define TILE_B 16
#define NTHREADS 256

// workspace layout in halves (f16), all section offsets multiple of 8 (16B aligned)
#define WS_RW1T 0                         // S x 64(N) x 128(K)   router W1 (first 128 rows), transposed
#define WS_W1T  (WS_RW1T + 3*64*128)      // S x T x 128(N) x 128(K)  expert W1 transposed
#define WS_W2T  (WS_W1T + 3*4*128*128)    // S x T x 64(N) x 128(K)   expert W2 transposed
#define WS_O1T  (WS_W2T + 3*4*64*128)     // 64(N) x 64(K)            head W1 transposed
#define WS_TOTAL (WS_O1T + 64*64)

// dynamic LDS layout (bytes)
#define SM_VALS 0          // 16*8*64 f16 = 16384 B
#define SM_P    16384      // 128*128 f16 = 32768 B
#define SM_HR   49152      // 128*64 f16  = 16384 B
#define SM_HT   65536      // 64*128 f16  = 16384 B (also reused as 128*64 head hidden)
#define SM_WB   81920      // 128*128 f16 = 32768 B (router W1 / expert W1 / head W1)
#define SM_W2B  114688     // 64*128 f16  = 16384 B
#define SM_POSH 131072     // 4*64 f32    = 1024 B
#define SM_IDX  132096     // 128 int     = 512 B
#define SMEM_BYTES 132608

// ---- branchless GELU (tanh approximation; V_TANH_F32 confirmed on gfx1250) ----
__device__ __forceinline__ float fast_tanh(float z) {
#if __has_builtin(__builtin_amdgcn_tanhf)
    return __builtin_amdgcn_tanhf(z);
#elif __has_builtin(__builtin_amdgcn_tanh_f32)
    return __builtin_amdgcn_tanh_f32(z);
#else
    float e = __expf(2.0f * z);
    return 1.0f - 2.0f / (e + 1.0f);
#endif
}
__device__ __forceinline__ float gelu_f(float x) {
    float x3 = x * x * x;
    return 0.5f * x * (1.0f + fast_tanh(0.7978845608028654f * (x + 0.044715f * x3)));
}

// ---- CDNA5 async global->LDS copy (ASYNCcnt path) ----
__device__ __forceinline__ void async_b128(void* lds_dst, const void* gsrc) {
    uint32_t l = (uint32_t)(uintptr_t)lds_dst;   // generic LDS addr: low 32 bits = LDS offset
    asm volatile("global_load_async_to_lds_b128 %0, %1, off"
                 :: "v"(l), "v"(gsrc) : "memory");
}
__device__ __forceinline__ void wait_async0() {
    asm volatile("s_wait_asynccnt 0" ::: "memory");
}
__device__ __forceinline__ void copy_g2s_async(_Float16* dst, const _Float16* src, int nhalves) {
    int n = nhalves >> 3;                        // 16B chunks
    for (int i = threadIdx.x; i < n; i += NTHREADS)
        async_b128((char*)dst + i * 16, (const char*)src + i * 16);
}

// Load a 16x32 f16 fragment (A or B) from row-major LDS [row][stride].
// CDNA5 wave32 layout (ISA 7.12.2): lane<16 -> row=lane, K={k0..k0+7, k0+16..k0+23}
//                                   lane>=16 -> row=lane-16, K shifted by +8.
__device__ __forceinline__ v16h load_frag(const _Float16* base, int stride, int row0, int k0) {
    int lane = threadIdx.x & 31;
    int r = row0 + (lane & 15);
    int k = k0 + ((lane >> 4) << 3);
    const _Float16* p = base + r * stride + k;
    v8h lo = *(const v8h*)(p);
    v8h hi = *(const v8h*)(p + 16);
    v16h o;
#pragma unroll
    for (int i = 0; i < 8; ++i) { o[i] = lo[i]; o[8 + i] = hi[i]; }
    return o;
}

// ---------------- prep: pack weights to f16, transposed [N][K] ----------------
__global__ void trix_pack(const float* __restrict__ rW1, const float* __restrict__ tW1,
                          const float* __restrict__ tW2, const float* __restrict__ oW1,
                          _Float16* __restrict__ ws) {
    int i = blockIdx.x * blockDim.x + threadIdx.x;
    if (i >= WS_TOTAL) return;
    float v;
    if (i < WS_W1T) {                       // router W1: [s][n][k] = rW1[s][k][n], k<128 of 136
        int s = i >> 13; int rem = i & 8191; int n = rem >> 7; int k = rem & 127;
        v = rW1[s * 136 * 64 + k * 64 + n];
    } else if (i < WS_W2T) {                // expert W1: [st][n][k] = tW1[st][k][n]
        int j = i - WS_W1T; int st = j >> 14; int rem = j & 16383; int n = rem >> 7; int k = rem & 127;
        v = tW1[(st * 128 + k) * 128 + n];
    } else if (i < WS_O1T) {                // expert W2: [st][n][k] = tW2[st][k][n]
        int j = i - WS_W2T; int st = j >> 13; int rem = j & 8191; int n = rem >> 7; int k = rem & 127;
        v = tW2[(st * 128 + k) * 64 + n];
    } else {                                // head W1: [n][k] = oW1[k][n]
        int j = i - WS_O1T; int n = j >> 6; int k = j & 63;
        v = oW1[k * 64 + n];
    }
    ws[i] = (_Float16)v;
}

// ---------------- fused forward: 16 batch elements per block ----------------
__global__ void __launch_bounds__(NTHREADS)
trix_main(const float* __restrict__ x,
          const float* __restrict__ embW, const float* __restrict__ embB,
          const float* __restrict__ lng,  const float* __restrict__ lnb,
          const float* __restrict__ posE, const float* __restrict__ stgE,
          const float* __restrict__ rW1,  const float* __restrict__ rb1,
          const float* __restrict__ rW2,  const float* __restrict__ rb2,
          const float* __restrict__ tb1,  const float* __restrict__ tb2,
          const float* __restrict__ ob1,  const float* __restrict__ oW2,
          const float* __restrict__ ob2,
          const _Float16* __restrict__ ws,
          float* __restrict__ out, int Bsz) {
    extern __shared__ __align__(16) char smem[];
    _Float16* vals = (_Float16*)(smem + SM_VALS);   // [16][8][64] f16
    _Float16* P    = (_Float16*)(smem + SM_P);      // [128][128]  rows 0-63 pair, 64-127 pair_flip
    _Float16* Hr   = (_Float16*)(smem + SM_HR);     // [128][64]   router hidden
    _Float16* Ht   = (_Float16*)(smem + SM_HT);     // [64][128]   expert hidden / [128][64] head hidden
    _Float16* Wb   = (_Float16*)(smem + SM_WB);     // current B-matrix (transposed [N][K])
    _Float16* W2b  = (_Float16*)(smem + SM_W2B);    // expert W2 (transposed [64][128])
    float*    posh = (float*)(smem + SM_POSH);      // [4][64] pos contribution + rb1 folded
    int*      idxA = (int*)(smem + SM_IDX);         // [128] routing argmax

    const int tid  = threadIdx.x;
    const int lane = tid & 31;
    const int wave = tid >> 5;
    const int b0   = blockIdx.x * TILE_B;

    // ---------------- Phase 0: Fourier features + embedding + layernorm ----------------
    if (tid < 128) {
        int bl = tid >> 3, v = tid & 7;
        int bg = b0 + bl;
        float e[64];
        if (bg < Bsz) {
            float xv = x[bg * NV_ + v];
            float ang0 = xv * 0.39269908169872414f;   // 2*pi/16
            float xf[12];
            float f = 1.0f;
#pragma unroll
            for (int q = 0; q < 6; ++q) { float a = ang0 * f; xf[q] = __sinf(a); xf[6 + q] = __cosf(a); f *= 2.0f; }
#pragma unroll 4
            for (int j = 0; j < 64; ++j) {
                float acc = embB[j];
#pragma unroll
                for (int i = 0; i < 12; ++i) acc += xf[i] * embW[i * 64 + j];
                e[j] = acc;
            }
            float mu = 0.f;
#pragma unroll 8
            for (int j = 0; j < 64; ++j) mu += e[j];
            mu *= (1.0f / 64.0f);
            float var = 0.f;
#pragma unroll 8
            for (int j = 0; j < 64; ++j) { float d = e[j] - mu; var += d * d; }
            var *= (1.0f / 64.0f);
            float inv = rsqrtf(var + 1e-5f);
#pragma unroll 4
            for (int j = 0; j < 64; ++j)
                vals[(bl * 8 + v) * 64 + j] = (_Float16)((e[j] - mu) * inv * lng[j] + lnb[j]);
        } else {
#pragma unroll 4
            for (int j = 0; j < 64; ++j) vals[(bl * 8 + v) * 64 + j] = (_Float16)0.0f;
        }
    }
    __syncthreads();

    // ---------------- Stages ----------------
    for (int s = 0; s < 3; ++s) {
        const int stride = 1 << s;

        // issue router W1 async copy; overlaps P build + posh compute
        copy_g2s_async(Wb, ws + WS_RW1T + s * 8192, 8192);

        // Build P: rows 0-63 = [a|b], rows 64-127 = [b|a], in 16-byte segments
        for (int q = tid; q < 128 * 16; q += NTHREADS) {
            int r = q >> 4, seg = q & 15;
            int rr = r & 63;
            int bl = rr >> 2, p = rr & 3;
            int lo = p & (stride - 1), hi = p & ~(stride - 1);
            int ia = (hi << 1) | lo, ib = ia | stride;
            bool firstHalf = (seg < 8);
            bool flip = (r >= 64);
            int node = (firstHalf != flip) ? ia : ib;
            int segin = firstHalf ? seg : seg - 8;
            *(uint4*)(P + r * 128 + seg * 8) =
                *(const uint4*)(vals + (bl * 8 + node) * 64 + segin * 8);
        }
        // posh[p][j] = pos-part of router hidden + rb1 folded
        {
            int p = tid >> 6, j = tid & 63;
            int lo = p & (stride - 1), hi = p & ~(stride - 1);
            int ia = (hi << 1) | lo;
            float acc = rb1[s * 64 + j];
#pragma unroll
            for (int q = 0; q < 4; ++q) {
                acc += posE[ia * 4 + q] * rW1[s * 8704 + (128 + q) * 64 + j];
                acc += stgE[s * 4 + q]  * rW1[s * 8704 + (132 + q) * 64 + j];
            }
            posh[p * 64 + j] = acc;
        }
        wait_async0();
        __syncthreads();

        // Router hidden: Hr(128x64) = gelu(P @ rW1[:128] + posh)
        // per-wave: nt = wave&3 fixed -> hoist B fragments; mt = (wave>>2) + 2i
        {
            const int nt = wave & 3;
            v16h bf[4];
#pragma unroll
            for (int ks = 0; ks < 4; ++ks) bf[ks] = load_frag(Wb, 128, nt * 16, ks * 32);
#pragma unroll
            for (int i = 0; i < 4; ++i) {
                int mt = (wave >> 2) + 2 * i;
                v8f acc = {};
#pragma unroll
                for (int ks = 0; ks < 4; ++ks) {
                    v16h a = load_frag(P, 128, mt * 16, ks * 32);
                    acc = __builtin_amdgcn_wmma_f32_16x16x32_f16(false, a, false, bf[ks], (short)0, acc, false, false);
                }
                int j = nt * 16 + (lane & 15);
                int rbase = mt * 16 + ((lane >> 4) << 3);
#pragma unroll
                for (int v = 0; v < 8; ++v) {
                    int r = rbase + v;
                    Hr[r * 64 + j] = (_Float16)gelu_f(acc[v] + posh[(r & 3) * 64 + j]);
                }
            }
        }
        __syncthreads();

        // issue expert-0 weights; overlaps logits/argmax phase
        copy_g2s_async(Wb,  ws + WS_W1T + (s * 4) * 16384, 16384);
        copy_g2s_async(W2b, ws + WS_W2T + (s * 4) * 8192, 8192);

        // Logits + argmax (T=4, tiny -> VALU). TEMP>0 does not change argmax.
        if (tid < 128) {
            float lg[4];
#pragma unroll
            for (int c = 0; c < 4; ++c) lg[c] = rb2[s * 4 + c];
            const float* w2 = rW2 + s * 256;
#pragma unroll 4
            for (int j = 0; j < 64; ++j) {
                float hv = (float)Hr[tid * 64 + j];
#pragma unroll
                for (int c = 0; c < 4; ++c) lg[c] += hv * w2[j * 4 + c];
            }
            int best = 0;
            float bv = lg[0];
#pragma unroll
            for (int c = 1; c < 4; ++c) if (lg[c] > bv) { bv = lg[c]; best = c; }
            idxA[tid] = best;
        }

        // Experts: compute all 4 (as reference does), masked writeback via routing index.
        // Software pipeline: W1(t+1) issued after Ht releases Wb; W2(t+1) after Ot releases W2b.
        for (int t = 0; t < 4; ++t) {
            wait_async0();
            __syncthreads();   // weights for expert t present in all waves; idxA visible (t=0)

            // Ht(64x128) = gelu(pair @ tW1[t] + tb1)
            // per-wave: nt = wave fixed -> hoist B; mt = i
            const float* tb1g = tb1 + (s * 4 + t) * 128;
            {
                v16h bf[4];
#pragma unroll
                for (int ks = 0; ks < 4; ++ks) bf[ks] = load_frag(Wb, 128, wave * 16, ks * 32);
#pragma unroll
                for (int mt = 0; mt < 4; ++mt) {
                    v8f acc = {};
#pragma unroll
                    for (int ks = 0; ks < 4; ++ks) {
                        v16h a = load_frag(P, 128, mt * 16, ks * 32);
                        acc = __builtin_amdgcn_wmma_f32_16x16x32_f16(false, a, false, bf[ks], (short)0, acc, false, false);
                    }
                    int j = wave * 16 + (lane & 15);
                    int rbase = mt * 16 + ((lane >> 4) << 3);
                    float bj = tb1g[j];
#pragma unroll
                    for (int v = 0; v < 8; ++v)
                        Ht[(rbase + v) * 128 + j] = (_Float16)gelu_f(acc[v] + bj);
                }
            }
            __syncthreads();

            // prefetch next expert's W1 into Wb (no longer read this iteration)
            if (t < 3) copy_g2s_async(Wb, ws + WS_W1T + (s * 4 + t + 1) * 16384, 16384);

            // Ot(64x64) = Ht @ tW2[t] + tb2; write rows whose route picked expert t
            // per-wave: nt = wave&3 fixed -> hoist B; mt = (wave>>2) + 2i
            const float* tb2g = tb2 + (s * 4 + t) * 64;
            {
                const int nt = wave & 3;
                v16h bf[4];
#pragma unroll
                for (int ks = 0; ks < 4; ++ks) bf[ks] = load_frag(W2b, 128, nt * 16, ks * 32);
#pragma unroll
                for (int i = 0; i < 2; ++i) {
                    int mt = (wave >> 2) + 2 * i;
                    v8f acc = {};
#pragma unroll
                    for (int ks = 0; ks < 4; ++ks) {
                        v16h a = load_frag(Ht, 128, mt * 16, ks * 32);
                        acc = __builtin_amdgcn_wmma_f32_16x16x32_f16(false, a, false, bf[ks], (short)0, acc, false, false);
                    }
                    int j = nt * 16 + (lane & 15);
                    int rbase = mt * 16 + ((lane >> 4) << 3);
                    float bj = tb2g[j];
#pragma unroll
                    for (int v = 0; v < 8; ++v) {
                        int r = rbase + v;
                        int bl = r >> 2, p = r & 3;
                        int lo = p & (stride - 1), hi = p & ~(stride - 1);
                        int ia = (hi << 1) | lo, ib = ia | stride;
                        float val = acc[v] + bj;
                        if (idxA[r] == t)      vals[(bl * 8 + ia) * 64 + j] = (_Float16)val;
                        if (idxA[64 + r] == t) vals[(bl * 8 + ib) * 64 + j] = (_Float16)val;
                    }
                }
            }
            __syncthreads();

            // prefetch next expert's W2 into W2b
            if (t < 3) copy_g2s_async(W2b, ws + WS_W2T + (s * 4 + t + 1) * 8192, 8192);
        }
    }

    // ---------------- Output head: Hh(128x64) = gelu(vals @ oW1 + ob1) ----------------
    copy_g2s_async(Wb, ws + WS_O1T, 4096);   // oW1 transposed [64][64]
    wait_async0();
    __syncthreads();
    {
        const int nt = wave & 3;
        v16h bf[2];
#pragma unroll
        for (int ks = 0; ks < 2; ++ks) bf[ks] = load_frag(Wb, 64, nt * 16, ks * 32);
#pragma unroll
        for (int i = 0; i < 4; ++i) {
            int mt = (wave >> 2) + 2 * i;
            v8f acc = {};
#pragma unroll
            for (int ks = 0; ks < 2; ++ks) {
                v16h a = load_frag(vals, 64, mt * 16, ks * 32);
                acc = __builtin_amdgcn_wmma_f32_16x16x32_f16(false, a, false, bf[ks], (short)0, acc, false, false);
            }
            int j = nt * 16 + (lane & 15);
            int rbase = mt * 16 + ((lane >> 4) << 3);
            float bj = ob1[j];
#pragma unroll
            for (int v = 0; v < 8; ++v)
                Ht[(rbase + v) * 64 + j] = (_Float16)gelu_f(acc[v] + bj);
        }
    }
    __syncthreads();

    // out = Hh @ oW2 + ob2  (N=1 -> VALU dot)
    if (tid < 128) {
        int bl = tid >> 3, v = tid & 7;
        int bg = b0 + bl;
        if (bg < Bsz) {
            float acc = 0.f;
#pragma unroll 8
            for (int j = 0; j < 64; ++j) acc += (float)Ht[tid * 64 + j] * oW2[j];
            out[bg * NV_ + v] = acc + ob2[0];
        }
    }
}

extern "C" void kernel_launch(void* const* d_in, const int* in_sizes, int n_in,
                              void* d_out, int out_size, void* d_ws, size_t ws_size,
                              hipStream_t stream) {
    const float* x    = (const float*)d_in[0];
    const float* embW = (const float*)d_in[1];
    const float* embB = (const float*)d_in[2];
    const float* lng  = (const float*)d_in[3];
    const float* lnb  = (const float*)d_in[4];
    const float* posE = (const float*)d_in[5];
    const float* stgE = (const float*)d_in[6];
    const float* rW1  = (const float*)d_in[7];
    const float* rb1  = (const float*)d_in[8];
    const float* rW2  = (const float*)d_in[9];
    const float* rb2  = (const float*)d_in[10];
    const float* tW1  = (const float*)d_in[11];
    const float* tb1  = (const float*)d_in[12];
    const float* tW2  = (const float*)d_in[13];
    const float* tb2  = (const float*)d_in[14];
    const float* oW1  = (const float*)d_in[15];
    const float* ob1  = (const float*)d_in[16];
    const float* oW2  = (const float*)d_in[17];
    const float* ob2  = (const float*)d_in[18];
    _Float16* ws = (_Float16*)d_ws;
    float* out = (float*)d_out;

    int Bsz = in_sizes[0] / NV_;

    trix_pack<<<(WS_TOTAL + 255) / 256, 256, 0, stream>>>(rW1, tW1, tW2, oW1, ws);

    (void)hipFuncSetAttribute((const void*)trix_main,
                              hipFuncAttributeMaxDynamicSharedMemorySize, SMEM_BYTES);
    int nblocks = (Bsz + TILE_B - 1) / TILE_B;
    trix_main<<<nblocks, NTHREADS, SMEM_BYTES, stream>>>(
        x, embW, embB, lng, lnb, posE, stgE, rW1, rb1, rW2, rb2,
        tb1, tb2, ob1, oW2, ob2, ws, out, Bsz);
}
